// PointFlow2DVAE_32710470926336
// MI455X (gfx1250) — compile-verified, matrix operand order checked by reading, not run on hardware
//
#include <hip/hip_runtime.h>
#include <hip/hip_bf16.h>
#include <math.h>

typedef __bf16 bf16;
typedef __attribute__((ext_vector_type(16))) __bf16 v16bf;
typedef __attribute__((ext_vector_type(8)))  __bf16 v8bf;
typedef __attribute__((ext_vector_type(8)))  float  v8f;
typedef __attribute__((ext_vector_type(4)))  unsigned int u32x4;
typedef __attribute__((ext_vector_type(8)))  int i32x8;
typedef __attribute__((ext_vector_type(4)))  int i32x4;

#ifndef __has_builtin
#define __has_builtin(x) 0
#endif

#if __has_builtin(__builtin_amdgcn_tensor_load_to_lds) && \
    __has_builtin(__builtin_amdgcn_s_wait_tensorcnt)
#define HAVE_TDM 1
#else
#define HAVE_TDM 0
#endif

#if __has_builtin(__builtin_amdgcn_tanhf)
#define TANH(x) __builtin_amdgcn_tanhf(x)
#else
#define TANH(x) tanhf(x)
#endif

#define B_     8
#define N_     2048
#define LAT_   128
#define STEPS_ 10
#define IN_CNF 131
#define K1PAD  160
#define LOG2PI 1.8378770664093453f

// workspace layout (float offsets)
#define WS_G   0        // 8*256  pooled features
#define WS_Z   2048     // 8*128  latent z
#define WS_XR  3072     // 8*2048*2 generated points
#define WS_ACC 35840    // [0]=recon [1]=logdet [2]=prior [3]=entropy [4]=chamfer
#define WS_BW1 35856    // W1 padded bf16: 256*160 = 40960 bf16 = 20480 f
#define WS_BW2 56336    // W2 bf16: 65536 bf16 = 32768 f
#define WS_BE2 89104    // We2 bf16: 65536 bf16 = 32768 f  (end 121872 f)

// --------------------------------------------------------------------------
// WMMA helpers (CDNA5 wave32, 16x16x32 bf16 -> f32)
// --------------------------------------------------------------------------
__device__ __forceinline__ v16bf load_frag(const bf16* p) {
  // p = base + row*stride + kchunk*32 + half*8 ; chunks at +0 and +16
  v8bf lo = *(const v8bf*)p;
  v8bf hi = *(const v8bf*)(p + 16);
  v16bf r;
#pragma unroll
  for (int i = 0; i < 8; ++i) { r[i] = lo[i]; r[8 + i] = hi[i]; }
  return r;
}

__device__ __forceinline__ v8f wmma_bf16(v16bf a, v16bf b, v8f c) {
  return __builtin_amdgcn_wmma_f32_16x16x32_bf16(false, a, false, b,
                                                 (short)0, c, false, false);
}

// --------------------------------------------------------------------------
// TDM: 1-D bulk copy global bf16 -> LDS (data_size = 8B granules)
// --------------------------------------------------------------------------
#if HAVE_TDM
__device__ __forceinline__ void tdm_copy_to_lds(const void* gsrc,
                                                unsigned int lds_off,
                                                unsigned int n_qwords) {
  unsigned long long ga = (unsigned long long)gsrc;
  u32x4 g0;
  g0[0] = 1u;                                     // count=1, user mode
  g0[1] = lds_off;                                // lds_addr (bytes)
  g0[2] = (unsigned int)(ga & 0xffffffffull);     // global_addr[31:0]
  g0[3] = (unsigned int)((ga >> 32) & 0x01ffffffull) | (2u << 30); // type=2
  i32x8 g1;
  g1[0] = (3 << 16);                              // data_size = 8 bytes
  g1[1] = (int)((n_qwords & 0xffffu) << 16);      // tensor_dim0[15:0]
  g1[2] = (int)(((n_qwords >> 16) & 0xffffu) | (1u << 16)); // dim0 hi | dim1=1
  g1[3] = (int)((n_qwords & 0xffffu) << 16);      // tile_dim0
  g1[4] = 0;                                      // tile_dim1/2 unused
  g1[5] = (int)n_qwords;                          // tensor_dim0_stride lo
  g1[6] = 0;
  g1[7] = 0;
#if __clang_major__ >= 23
  __builtin_amdgcn_tensor_load_to_lds(g0, g1, (i32x4)0, (i32x4)0, (i32x8)0, 0);
#else
  __builtin_amdgcn_tensor_load_to_lds(g0, g1, (i32x4)0, (i32x4)0, 0);
#endif
}
#endif

// --------------------------------------------------------------------------
// init workspace accumulators / pooled-feature buffer
// --------------------------------------------------------------------------
__global__ void init_ws(float* ws) {
  int i = blockIdx.x * blockDim.x + threadIdx.x;
  if (i < 2048) ws[WS_G + i] = 0.f;
  if (i < 8)    ws[WS_ACC + i] = 0.f;
}

// --------------------------------------------------------------------------
// one-time f32 -> bf16 weight conversion (amortized over 2048+ tile blocks)
// --------------------------------------------------------------------------
__global__ void convert_weights(const float* __restrict__ W1,
                                const float* __restrict__ W2,
                                const float* __restrict__ We2,
                                float* ws) {
  bf16* bW1  = (bf16*)(ws + WS_BW1);
  bf16* bW2  = (bf16*)(ws + WS_BW2);
  bf16* bWe2 = (bf16*)(ws + WS_BE2);
  int i = blockIdx.x * blockDim.x + threadIdx.x;
  if (i < 256 * K1PAD) {
    int n = i / K1PAD, k = i % K1PAD;
    bW1[i] = (k < IN_CNF) ? (bf16)W1[n * IN_CNF + k] : (bf16)0.f;
  }
  if (i < 65536) {
    bW2[i]  = (bf16)W2[i];
    bWe2[i] = (bf16)We2[i];
  }
}

// --------------------------------------------------------------------------
// PointNet encoder: h1 = relu(x@We1^T+be1); h2 = relu(h1@We2^T+be2); maxpool
// 1024 blocks x 128 threads, 16 points per block (tiles never straddle batch)
// --------------------------------------------------------------------------
__global__ void encoder_kernel(const float* __restrict__ x,
                               const float* __restrict__ We1,
                               const float* __restrict__ be1,
                               const float* __restrict__ be2,
                               float* ws) {
  extern __shared__ char smem[];
  bf16*  sW2 = (bf16*)smem;                       // 256*256 bf16
  bf16*  sH1 = (bf16*)(smem + 131072);            // 16*256 bf16
  float* sB2 = (float*)(smem + 131072 + 8192);    // 256 f32

  const int tid  = threadIdx.x;
  const int lane = tid & 31, wave = tid >> 5;
  const int row  = lane & 15, half = lane >> 4;
  const int p0   = blockIdx.x * 16;
  const int b    = p0 / N_;

  const bf16* bWe2 = (const bf16*)(ws + WS_BE2);
  __builtin_prefetch(bWe2, 0, 0);

#if HAVE_TDM
  if (wave == 0) {
    tdm_copy_to_lds(bWe2, 0u /* sW2 LDS offset */, 16384u);
    __builtin_amdgcn_s_wait_tensorcnt(0);
  }
#else
  for (int i = tid; i < 8192; i += 128)
    ((u32x4*)sW2)[i] = ((const u32x4*)bWe2)[i];
#endif
  for (int i = tid; i < 256; i += 128) sB2[i] = be2[i];

  for (int i = tid; i < 16 * 256; i += 128) {
    int m = i >> 8, k = i & 255;
    float x0 = x[(p0 + m) * 2 + 0], x1 = x[(p0 + m) * 2 + 1];
    float h = x0 * We1[k * 2 + 0] + x1 * We1[k * 2 + 1] + be1[k];
    sH1[m * 256 + k] = (bf16)fmaxf(h, 0.f);
  }
  __syncthreads();

  v8f acc[4] = {};
#pragma unroll
  for (int c = 0; c < 8; ++c) {
    v16bf a = load_frag(sH1 + row * 256 + c * 32 + half * 8);
#pragma unroll
    for (int t = 0; t < 4; ++t) {
      int n0 = wave * 64 + t * 16;
      v16bf bb = load_frag(sW2 + (n0 + row) * 256 + c * 32 + half * 8);
      acc[t] = wmma_bf16(a, bb, acc[t]);
    }
  }
#pragma unroll
  for (int t = 0; t < 4; ++t) {
    int n = wave * 64 + t * 16 + row;
    float bb = sB2[n];
    float mx = 0.f;
#pragma unroll
    for (int r = 0; r < 8; ++r) mx = fmaxf(mx, fmaxf(acc[t][r] + bb, 0.f));
    mx = fmaxf(mx, __shfl_xor(mx, 16, 32));   // combine both M-halves
    if (half == 0)
      atomicMax((unsigned int*)(ws + WS_G + b * 256 + n), __float_as_uint(mx));
  }
}

// --------------------------------------------------------------------------
// latent head: mu/logvar, reparameterize, prior+entropy losses (tiny)
// --------------------------------------------------------------------------
__global__ void latent_kernel(const float* __restrict__ eps,
                              const float* __restrict__ Wmu,
                              const float* __restrict__ bmu,
                              const float* __restrict__ Wlv,
                              const float* __restrict__ blv,
                              float* ws) {
  int t = threadIdx.x;
  float priorp = 0.f, entp = 0.f;
  for (int i = t; i < B_ * LAT_; i += 256) {
    int b = i / LAT_, l = i % LAT_;
    const float* g = ws + WS_G + b * 256;
    float mu = bmu[l], lv = blv[l];
    for (int k = 0; k < 256; ++k) {
      mu += g[k] * Wmu[l * 256 + k];
      lv += g[k] * Wlv[l * 256 + k];
    }
    ws[WS_Z + i] = mu + eps[b * LAT_ + l] * expf(0.5f * lv);
    priorp += 0.5f * (mu * mu + expf(lv) - lv - 1.f);
    entp   += -0.5f * (lv + 1.f + LOG2PI);
  }
  atomicAdd(ws + WS_ACC + 2, priorp);
  atomicAdd(ws + WS_ACC + 3, entp);
}

// --------------------------------------------------------------------------
// CNF integration. FLOW=true: x->gaussian w/ exact divergence (2 JVPs whose
// input tangents reduce to columns 0/1 of W1). FLOW=false: generation branch.
// 1024 blocks x 128 threads (4 waves), 16 points/block, weights in LDS.
// --------------------------------------------------------------------------
#define S_W1   0          // 256*160 bf16 = 81920
#define S_W2   81920      // 256*256 bf16 = 131072
#define S_INP  212992     // 16*160 bf16 = 5120
#define S_HA   218112     // 16*256 bf16 = 8192
#define S_T0A  226304     // 8192
#define S_T1A  234496     // 8192
#define S_B1   242688     // 256 f32
#define S_B2   243712
#define S_W1X  244736
#define S_W1Y  245760
#define S_W3   246784     // 2*256 f32
#define S_B3   248832
#define S_Y    248896     // 16*2 f32
#define S_LD   249024     // 16 f32
#define S_VAC  249088     // 3*16 f32
#define SMEM_CNF 249344

template <bool FLOW>
__global__ void cnf_kernel(const float* __restrict__ xin,
                           const float* __restrict__ W1, const float* __restrict__ b1,
                           const float* __restrict__ b2,
                           const float* __restrict__ W3, const float* __restrict__ b3,
                           float* ws) {
  extern __shared__ char smem[];
  bf16*  sW1  = (bf16*)(smem + S_W1);
  bf16*  sW2  = (bf16*)(smem + S_W2);
  bf16*  sInp = (bf16*)(smem + S_INP);
  bf16*  sHA  = (bf16*)(smem + S_HA);
  bf16*  sT0  = (bf16*)(smem + S_T0A);
  bf16*  sT1  = (bf16*)(smem + S_T1A);
  float* sB1  = (float*)(smem + S_B1);
  float* sB2  = (float*)(smem + S_B2);
  float* sW1x = (float*)(smem + S_W1X);
  float* sW1y = (float*)(smem + S_W1Y);
  float* sW3  = (float*)(smem + S_W3);
  float* sB3  = (float*)(smem + S_B3);
  float* sY   = (float*)(smem + S_Y);
  float* sLD  = (float*)(smem + S_LD);
  float* sV0  = (float*)(smem + S_VAC);
  float* sV1  = sV0 + 16;
  float* sDV  = sV0 + 32;

  const int tid  = threadIdx.x;
  const int lane = tid & 31, wave = tid >> 5;
  const int row  = lane & 15, half = lane >> 4;
  const int p0   = blockIdx.x * 16;
  const int b    = p0 / N_;
  const float dt = 1.0f / STEPS_;

  const bf16* bW1 = (const bf16*)(ws + WS_BW1);
  const bf16* bW2 = (const bf16*)(ws + WS_BW2);
  __builtin_prefetch(bW1, 0, 0);
  __builtin_prefetch(bW2, 0, 0);

  // stage pre-converted bf16 weights into LDS
  for (int i = tid; i < 5120; i += 128)                 // W1: 80KB
    ((u32x4*)sW1)[i] = ((const u32x4*)bW1)[i];
#if HAVE_TDM
  if (wave == 0) {                                       // W2: 128KB via TDM
    tdm_copy_to_lds(bW2, (unsigned int)S_W2, 16384u);
    __builtin_amdgcn_s_wait_tensorcnt(0);
  }
#else
  for (int i = tid; i < 8192; i += 128)
    ((u32x4*)sW2)[i] = ((const u32x4*)bW2)[i];
#endif
  for (int i = tid; i < 256; i += 128) {
    sB1[i]  = b1[i];  sB2[i] = b2[i];
    sW1x[i] = W1[i * IN_CNF + 0];
    sW1y[i] = W1[i * IN_CNF + 1];
    sW3[i]  = W3[i];  sW3[256 + i] = W3[256 + i];
  }
  if (tid < 2) sB3[tid] = b3[tid];
  if (tid < 16) {
    sY[tid * 2 + 0] = xin[(p0 + tid) * 2 + 0];
    sY[tid * 2 + 1] = xin[(p0 + tid) * 2 + 1];
    sLD[tid] = 0.f; sV0[tid] = 0.f; sV1[tid] = 0.f; sDV[tid] = 0.f;
  }
  const float* z = ws + WS_Z + b * LAT_;
  __syncthreads();

#pragma unroll 1
  for (int step = 0; step < STEPS_; ++step) {
    const float tval = FLOW ? (step * dt) : (1.0f - step * dt);
    for (int i = tid; i < 16 * K1PAD; i += 128) {
      int m = i / K1PAD, k = i % K1PAD;
      float v;
      if (k < 2)            v = sY[m * 2 + k];
      else if (k == 2)      v = tval;
      else if (k < IN_CNF)  v = z[k - 3];
      else                  v = 0.f;
      sInp[i] = (bf16)v;
    }
    __syncthreads();

    // ---- layer 1: [16x160] @ W1^T -> [16x256] -------------------------
    v8f a1[4] = {};
#pragma unroll
    for (int c = 0; c < 5; ++c) {
      v16bf af = load_frag(sInp + row * K1PAD + c * 32 + half * 8);
#pragma unroll
      for (int t = 0; t < 4; ++t) {
        int n0 = wave * 64 + t * 16;
        v16bf wf = load_frag(sW1 + (n0 + row) * K1PAD + c * 32 + half * 8);
        a1[t] = wmma_bf16(af, wf, a1[t]);
      }
    }
#pragma unroll
    for (int t = 0; t < 4; ++t) {
      int n = wave * 64 + t * 16 + row;
      float bb = sB1[n], wx = sW1x[n], wy = sW1y[n];
#pragma unroll
      for (int r = 0; r < 8; ++r) {
        int m = r + 8 * half;
        float h = TANH(a1[t][r] + bb);
        sHA[m * 256 + n] = (bf16)h;
        if (FLOW) {
          float u = 1.f - h * h;              // tanh'
          sT0[m * 256 + n] = (bf16)(u * wx);  // JVP tangent dir e0
          sT1[m * 256 + n] = (bf16)(u * wy);  // JVP tangent dir e1
        }
      }
    }
    __syncthreads();

    // ---- layer 2 forward + two tangent GEMMs (shared B fragments) -----
    v8f a2[4] = {}, d0[4] = {}, d1[4] = {};
#pragma unroll
    for (int c = 0; c < 8; ++c) {
      v16bf af = load_frag(sHA + row * 256 + c * 32 + half * 8);
      v16bf t0f = {}, t1f = {};
      if (FLOW) {
        t0f = load_frag(sT0 + row * 256 + c * 32 + half * 8);
        t1f = load_frag(sT1 + row * 256 + c * 32 + half * 8);
      }
#pragma unroll
      for (int t = 0; t < 4; ++t) {
        int n0 = wave * 64 + t * 16;
        v16bf wf = load_frag(sW2 + (n0 + row) * 256 + c * 32 + half * 8);
        a2[t] = wmma_bf16(af, wf, a2[t]);
        if (FLOW) {
          d0[t] = wmma_bf16(t0f, wf, d0[t]);
          d1[t] = wmma_bf16(t1f, wf, d1[t]);
        }
      }
    }

    // ---- output layer (256->2) + divergence, shuffle-reduce over N ----
#pragma unroll
    for (int t = 0; t < 4; ++t) {
      int n = wave * 64 + t * 16 + row;
      float bb = sB2[n], w30 = sW3[n], w31 = sW3[256 + n];
#pragma unroll
      for (int r = 0; r < 8; ++r) {
        int m = r + 8 * half;
        float h2 = TANH(a2[t][r] + bb);
        float pv0 = h2 * w30, pv1 = h2 * w31, pdv = 0.f;
        if (FLOW) {
          float u = 1.f - h2 * h2;
          pdv = (u * d0[t][r]) * w30 + (u * d1[t][r]) * w31;
        }
#pragma unroll
        for (int s = 1; s < 16; s <<= 1) {
          pv0 += __shfl_xor(pv0, s, 32);
          pv1 += __shfl_xor(pv1, s, 32);
          if (FLOW) pdv += __shfl_xor(pdv, s, 32);
        }
        if (row == 0) {
          atomicAdd(&sV0[m], pv0);
          atomicAdd(&sV1[m], pv1);
          if (FLOW) atomicAdd(&sDV[m], pdv);
        }
      }
    }
    __syncthreads();

    if (tid < 16) {
      float v0 = sV0[tid] + sB3[0];
      float v1 = sV1[tid] + sB3[1];
      if (FLOW) {
        sY[tid * 2 + 0] += dt * v0;
        sY[tid * 2 + 1] += dt * v1;
        sLD[tid] += dt * sDV[tid];
      } else {
        sY[tid * 2 + 0] -= dt * v0;
        sY[tid * 2 + 1] -= dt * v1;
      }
      sV0[tid] = 0.f; sV1[tid] = 0.f; sDV[tid] = 0.f;
    }
    __syncthreads();
  }

  if (tid < 16) {
    if (FLOW) {
      float y0 = sY[tid * 2], y1 = sY[tid * 2 + 1], ld = sLD[tid];
      float log_py = -0.5f * (y0 * y0 + LOG2PI) - 0.5f * (y1 * y1 + LOG2PI);
      atomicAdd(ws + WS_ACC + 0, -(log_py + ld));
      atomicAdd(ws + WS_ACC + 1, ld);
    } else {
      float* xr = ws + WS_XR;
      xr[(p0 + tid) * 2 + 0] = sY[tid * 2 + 0];
      xr[(p0 + tid) * 2 + 1] = sY[tid * 2 + 1];
    }
  }
}

// --------------------------------------------------------------------------
// chamfer: both directional mins in one pass; LDS-tiled j loop
// --------------------------------------------------------------------------
__global__ void chamfer_kernel(const float* __restrict__ x, float* ws) {
  __shared__ float sx[256 * 2];
  __shared__ float sxr[256 * 2];
  const float* xr = ws + WS_XR;
  int b = blockIdx.x >> 4;
  int i = (blockIdx.x & 15) * 128 + threadIdx.x;
  float xi0 = x[(b * N_ + i) * 2 + 0],  xi1 = x[(b * N_ + i) * 2 + 1];
  float ri0 = xr[(b * N_ + i) * 2 + 0], ri1 = xr[(b * N_ + i) * 2 + 1];
  float min1 = 3.0e38f, min2 = 3.0e38f;
  for (int j0 = 0; j0 < N_; j0 += 256) {
    __syncthreads();
    for (int k = threadIdx.x; k < 256; k += 128) {
      sx[k * 2 + 0]  = x[(b * N_ + j0 + k) * 2 + 0];
      sx[k * 2 + 1]  = x[(b * N_ + j0 + k) * 2 + 1];
      sxr[k * 2 + 0] = xr[(b * N_ + j0 + k) * 2 + 0];
      sxr[k * 2 + 1] = xr[(b * N_ + j0 + k) * 2 + 1];
    }
    __syncthreads();
    for (int k = 0; k < 256; ++k) {
      float dx = ri0 - sx[k * 2], dy = ri1 - sx[k * 2 + 1];
      min1 = fminf(min1, sqrtf(dx * dx + dy * dy));
      float ex = xi0 - sxr[k * 2], ey = xi1 - sxr[k * 2 + 1];
      min2 = fminf(min2, sqrtf(ex * ex + ey * ey));
    }
  }
  atomicAdd(ws + WS_ACC + 4, min1 + min2);
}

// --------------------------------------------------------------------------
__global__ void combine_kernel(const float* ws, float* out) {
  const float inv_bn = 1.0f / (B_ * N_);
  float recon = ws[WS_ACC + 0] * inv_bn;
  float ldm   = ws[WS_ACC + 1] * inv_bn;
  float prior = ws[WS_ACC + 2] / B_;
  float ent   = ws[WS_ACC + 3] / B_;
  float cham  = ws[WS_ACC + 4] * inv_bn;
  float vol   = fmaxf(ldm - 10.f, 0.f);
  out[0] = 1.0f * recon + 0.1f * prior + 0.01f * ent + 10.0f * cham + 0.01f * vol;
}

// --------------------------------------------------------------------------
extern "C" void kernel_launch(void* const* d_in, const int* in_sizes, int n_in,
                              void* d_out, int out_size, void* d_ws, size_t ws_size,
                              hipStream_t stream) {
  (void)in_sizes; (void)n_in; (void)out_size; (void)ws_size;
  const float* x    = (const float*)d_in[0];
  const float* eps  = (const float*)d_in[1];
  const float* noise= (const float*)d_in[2];
  const float* We1  = (const float*)d_in[3];
  const float* be1  = (const float*)d_in[4];
  const float* We2  = (const float*)d_in[5];
  const float* be2  = (const float*)d_in[6];
  const float* Wmu  = (const float*)d_in[7];
  const float* bmu  = (const float*)d_in[8];
  const float* Wlv  = (const float*)d_in[9];
  const float* blv  = (const float*)d_in[10];
  const float* W1   = (const float*)d_in[11];
  const float* b1   = (const float*)d_in[12];
  const float* W2   = (const float*)d_in[13];
  const float* b2   = (const float*)d_in[14];
  const float* W3   = (const float*)d_in[15];
  const float* b3   = (const float*)d_in[16];
  float* ws  = (float*)d_ws;
  float* out = (float*)d_out;

  hipLaunchKernelGGL(init_ws, dim3(8), dim3(256), 0, stream, ws);
  hipLaunchKernelGGL(convert_weights, dim3(256), dim3(256), 0, stream,
                     W1, W2, We2, ws);
  hipLaunchKernelGGL(encoder_kernel, dim3(1024), dim3(128), 140288, stream,
                     x, We1, be1, be2, ws);
  hipLaunchKernelGGL(latent_kernel, dim3(1), dim3(256), 0, stream,
                     eps, Wmu, bmu, Wlv, blv, ws);
  hipLaunchKernelGGL((cnf_kernel<true>), dim3(1024), dim3(128), SMEM_CNF, stream,
                     x, W1, b1, b2, W3, b3, ws);
  hipLaunchKernelGGL((cnf_kernel<false>), dim3(1024), dim3(128), SMEM_CNF, stream,
                     noise, W1, b1, b2, W3, b3, ws);
  hipLaunchKernelGGL(chamfer_kernel, dim3(128), dim3(128), 0, stream, x, ws);
  hipLaunchKernelGGL(combine_kernel, dim3(1), dim3(1), 0, stream, ws, out);
}
